// SupervisedContrastiveLoss_43353399885893
// MI455X (gfx1250) — matrix operand order repaired and checked by reading, not compiled, not run on hardware
//
#include <hip/hip_runtime.h>
#include <hip/hip_bf16.h>

typedef _Float16 v16h __attribute__((ext_vector_type(16)));
typedef _Float16 v8h  __attribute__((ext_vector_type(8)));
typedef float    v8f  __attribute__((ext_vector_type(8)));

#define DIM 256
#define CHUNKS 4
#define INV_TEMP 14.2857142857142857f  // 1/0.07

// ---------------- Kernel 1: row L2-normalize, emit f16 ----------------
__global__ __launch_bounds__(256)
void scl_normalize_kernel(const float* __restrict__ x, _Float16* __restrict__ hn) {
  __shared__ float red[256];
  const int row = blockIdx.x;
  const int k = threadIdx.x;
  float v = x[row * DIM + k];
  red[k] = v * v;
  __syncthreads();
  for (int s = 128; s > 0; s >>= 1) {
    if (k < s) red[k] += red[k + s];
    __syncthreads();
  }
  float nrm = fmaxf(sqrtf(red[0]), 1e-12f);
  hn[row * DIM + k] = (_Float16)(v / nrm);
}

// ---- helpers ----
__device__ __forceinline__ v8f wmma_tile(const v16h a[8], const v16h b[8]) {
  v8f c = {};
#pragma unroll
  for (int kk = 0; kk < 8; ++kk)
    c = __builtin_amdgcn_wmma_f32_16x16x32_f16(false, a[kk], false, b[kk],
                                               (short)0, c, false, false);
  return c;
}

// Load one 32x16 (KxN) B tile: lane = N column, half 0 -> K {0..7, 16..23}, half 1 -> K {8..15, 24..31}.
__device__ __forceinline__ void load_btile(const _Float16* __restrict__ hn,
                                           const int* __restrict__ labels,
                                           int colBase, int l15, int half,
                                           v16h b[8], int& labCol) {
  const _Float16* cp = hn + (colBase + l15) * DIM + (half ? 8 : 0);
#pragma unroll
  for (int kk = 0; kk < 8; ++kk) {
    v8h lo = *(const v8h*)(cp + kk * 32);
    v8h hi = *(const v8h*)(cp + kk * 32 + 16);
    v16h bv;
#pragma unroll
    for (int i = 0; i < 8; ++i) { bv[i] = lo[i]; bv[i + 8] = hi[i]; }
    b[kk] = bv;
  }
  labCol = labels[colBase + l15];
}

// Load resident A fragments: 16 rows x 256 K as 8 k-steps of 16x32 f16.
// ISA A layout: both lane-halves hold M=0..15; half 0 carries
// K = {kk*32+0..7, kk*32+16..23}, half 1 carries K = {+8..15, +24..31}.
__device__ __forceinline__ void load_atile(const _Float16* __restrict__ hn,
                                           int rowBase, int l15, int half,
                                           v16h a[8]) {
  const _Float16* rp = hn + (rowBase + l15) * DIM + (half ? 8 : 0);
#pragma unroll
  for (int kk = 0; kk < 8; ++kk) {
    const _Float16* p = rp + kk * 32;
    v8h lo = *(const v8h*)(p);
    v8h hi = *(const v8h*)(p + 16);
    v16h av;
#pragma unroll
    for (int i = 0; i < 8; ++i) { av[i] = lo[i]; av[i + 8] = hi[i]; }
    a[kk] = av;
  }
}

// Deterministic cross-lane reduce within each 16-lane half; lanes 0/16 then
// write the 8 rows they own (rows r + 8*half) to dst[row].
__device__ __forceinline__ void reduce_write_rows(float acc[8], int rowBase,
                                                  int l15, int half,
                                                  float* __restrict__ dst) {
#pragma unroll
  for (int r = 0; r < 8; ++r) {
    float v = acc[r];
    for (int m = 1; m < 16; m <<= 1) v += __shfl_xor(v, m, 32);
    acc[r] = v;
  }
  if (l15 == 0) {
#pragma unroll
    for (int r = 0; r < 8; ++r) dst[rowBase + r + 8 * half] = acc[r];
  }
}

// ---------------- Kernel 2a: negative partition sums ----------------
// Grid: (CHUNKS, nRows/64). WG = 4 waves; wave w owns rows rowBlock*64+w*16..+16.
// All 4 waves walk the SAME tile sequence -> B tiles fetched once per WGP
// (WGP$ reuse); periodic barrier keeps the waves converged on the stream.
__global__ __launch_bounds__(128)
void scl_neg_kernel(const _Float16* __restrict__ hn,
                    const int* __restrict__ labels,
                    float* __restrict__ negPart,   // [CHUNKS][nRows]
                    int nRows) {
  const int lane = threadIdx.x & 31;
  const int wave = threadIdx.x >> 5;
  const int half = lane >> 4;
  const int l15  = lane & 15;
  const int rowBase = blockIdx.y * 64 + wave * 16;
  const int nTiles = nRows >> 4;
  const int tpc = nTiles / CHUNKS;
  const int t0 = blockIdx.x * tpc;
  const int tEnd = t0 + tpc;

  v16h a[8];
  load_atile(hn, rowBase, l15, half, a);

  int labRow[8];
#pragma unroll
  for (int r = 0; r < 8; ++r) labRow[r] = labels[rowBase + r + 8 * half];

  float negAcc[8];
#pragma unroll
  for (int r = 0; r < 8; ++r) negAcc[r] = 0.f;

  if ((tpc & 1) == 0) {
    v16h bA[8], bB[8];
    int labA, labB;
    load_btile(hn, labels, t0 * 16, l15, half, bA, labA);
    int it = 0;
    for (int t = t0; t < tEnd; t += 2, ++it) {
      load_btile(hn, labels, (t + 1) * 16, l15, half, bB, labB);
      v8f c = wmma_tile(a, bA);
#pragma unroll
      for (int r = 0; r < 8; ++r)
        if (labA != labRow[r]) negAcc[r] += __expf(c[r] * INV_TEMP);
      const int tn = (t + 2 < tEnd) ? t + 2 : t0;
      load_btile(hn, labels, tn * 16, l15, half, bA, labA);
      c = wmma_tile(a, bB);
#pragma unroll
      for (int r = 0; r < 8; ++r)
        if (labB != labRow[r]) negAcc[r] += __expf(c[r] * INV_TEMP);
      if ((it & 7) == 7) __syncthreads();   // keep waves on the same tiles
    }
  } else {
    for (int t = t0; t < tEnd; ++t) {
      v16h b[8]; int lab;
      load_btile(hn, labels, t * 16, l15, half, b, lab);
      v8f c = wmma_tile(a, b);
#pragma unroll
      for (int r = 0; r < 8; ++r)
        if (lab != labRow[r]) negAcc[r] += __expf(c[r] * INV_TEMP);
    }
  }

  reduce_write_rows(negAcc, rowBase, l15, half, negPart + blockIdx.x * nRows);
}

// ---------------- Kernel 2b: positive pair losses ----------------
__global__ __launch_bounds__(128)
void scl_pos_kernel(const _Float16* __restrict__ hn,
                    const int* __restrict__ labels,
                    const float* __restrict__ negPart, // [CHUNKS][nRows]
                    float* __restrict__ posPart,       // [CHUNKS][nRows]
                    int*   __restrict__ cntPart,       // [CHUNKS][nRows]
                    int nRows) {
  const int lane = threadIdx.x & 31;
  const int wave = threadIdx.x >> 5;
  const int half = lane >> 4;
  const int l15  = lane & 15;
  const int rowBase = blockIdx.y * 64 + wave * 16;
  const int nTiles = nRows >> 4;
  const int tpc = nTiles / CHUNKS;
  const int t0 = blockIdx.x * tpc;
  const int tEnd = t0 + tpc;

  v16h a[8];
  load_atile(hn, rowBase, l15, half, a);

  int labRow[8];
  float negRow[8];
#pragma unroll
  for (int r = 0; r < 8; ++r) {
    const int row = rowBase + r + 8 * half;
    labRow[r] = labels[row];
    float s = 0.f;
#pragma unroll
    for (int ch = 0; ch < CHUNKS; ++ch) s += negPart[ch * nRows + row];
    negRow[r] = s;
  }

  float posAcc[8];
  int   cntAcc[8];
#pragma unroll
  for (int r = 0; r < 8; ++r) { posAcc[r] = 0.f; cntAcc[r] = 0; }

  if ((tpc & 1) == 0) {
    v16h bA[8], bB[8];
    int labA, labB;
    load_btile(hn, labels, t0 * 16, l15, half, bA, labA);
    int it = 0;
    for (int t = t0; t < tEnd; t += 2, ++it) {
      load_btile(hn, labels, (t + 1) * 16, l15, half, bB, labB);
      v8f c = wmma_tile(a, bA);
#pragma unroll
      for (int r = 0; r < 8; ++r) {
        const int i = rowBase + r + 8 * half;
        const int j = t * 16 + l15;
        if (labA == labRow[r] && j != i) {
          posAcc[r] += __logf(1.0f + negRow[r] * __expf(-c[r] * INV_TEMP));
          cntAcc[r] += 1;
        }
      }
      const int tn = (t + 2 < tEnd) ? t + 2 : t0;
      load_btile(hn, labels, tn * 16, l15, half, bA, labA);
      c = wmma_tile(a, bB);
#pragma unroll
      for (int r = 0; r < 8; ++r) {
        const int i = rowBase + r + 8 * half;
        const int j = (t + 1) * 16 + l15;
        if (labB == labRow[r] && j != i) {
          posAcc[r] += __logf(1.0f + negRow[r] * __expf(-c[r] * INV_TEMP));
          cntAcc[r] += 1;
        }
      }
      if ((it & 7) == 7) __syncthreads();
    }
  } else {
    for (int t = t0; t < tEnd; ++t) {
      v16h b[8]; int lab;
      load_btile(hn, labels, t * 16, l15, half, b, lab);
      v8f c = wmma_tile(a, b);
#pragma unroll
      for (int r = 0; r < 8; ++r) {
        const int i = rowBase + r + 8 * half;
        const int j = t * 16 + l15;
        if (lab == labRow[r] && j != i) {
          posAcc[r] += __logf(1.0f + negRow[r] * __expf(-c[r] * INV_TEMP));
          cntAcc[r] += 1;
        }
      }
    }
  }

  // count reduce (int) + loss reduce (float), deterministic, unique rows/wave
#pragma unroll
  for (int r = 0; r < 8; ++r) {
    int n = cntAcc[r];
    for (int m = 1; m < 16; m <<= 1) n += __shfl_xor(n, m, 32);
    cntAcc[r] = n;
  }
  if (l15 == 0) {
#pragma unroll
    for (int r = 0; r < 8; ++r)
      cntPart[blockIdx.x * nRows + rowBase + r + 8 * half] = cntAcc[r];
  }
  reduce_write_rows(posAcc, rowBase, l15, half, posPart + blockIdx.x * nRows);
}

// ---------------- Kernel 3: deterministic final reduction ----------------
__global__ __launch_bounds__(1024)
void scl_final_kernel(const float* __restrict__ posPart,
                      const int* __restrict__ cntPart,
                      float* __restrict__ out, int nRows) {
  __shared__ float sl[1024];
  __shared__ int   sv[1024];
  const int tid = threadIdx.x;
  float accL = 0.f;
  int accV = 0;
  for (int row = tid; row < nRows; row += 1024) {
    float p = 0.f;
    int c = 0;
#pragma unroll
    for (int ch = 0; ch < CHUNKS; ++ch) {
      p += posPart[ch * nRows + row];
      c += cntPart[ch * nRows + row];
    }
    if (c > 0) { accL += p / (float)c; accV += 1; }
  }
  sl[tid] = accL;
  sv[tid] = accV;
  __syncthreads();
  for (int s = 512; s > 0; s >>= 1) {
    if (tid < s) { sl[tid] += sl[tid + s]; sv[tid] += sv[tid + s]; }
    __syncthreads();
  }
  if (tid == 0) out[0] = (sv[0] > 0) ? sl[0] / (float)sv[0] : 0.f;
}

extern "C" void kernel_launch(void* const* d_in, const int* in_sizes, int n_in,
                              void* d_out, int out_size, void* d_ws, size_t ws_size,
                              hipStream_t stream) {
  const float* hv     = (const float*)d_in[0];
  const int*   labels = (const int*)d_in[1];
  float* out = (float*)d_out;

  const int N = in_sizes[0] / DIM;   // 8192

  char* ws = (char*)d_ws;
  // [CHUNKS][N] partial arrays, then the f16 normalized matrix.
  float*    negPart = (float*)ws;                                   // 128 KB
  float*    posPart = (float*)(ws + (size_t)CHUNKS * N * 4);        // 128 KB
  int*      cntPart = (int*)  (ws + (size_t)2 * CHUNKS * N * 4);    // 128 KB
  _Float16* hn      = (_Float16*)(ws + (size_t)3 * CHUNKS * N * 4); // 4 MB

  dim3 grid2(CHUNKS, N / 64);

  scl_normalize_kernel<<<N, 256, 0, stream>>>(hv, hn);
  scl_neg_kernel<<<grid2, 128, 0, stream>>>(hn, labels, negPart, N);
  scl_pos_kernel<<<grid2, 128, 0, stream>>>(hn, labels, negPart, posPart, cntPart, N);
  scl_final_kernel<<<1, 1024, 0, stream>>>(posPart, cntPart, out, N);
}
